// EdgeguidedRankingLoss_54760833024307
// MI455X (gfx1250) — matrix operand order repaired and checked by reading, not compiled, not run on hardware
//
#include <hip/hip_runtime.h>
#include <hip/hip_bf16.h>
#include <math.h>

#define NIMG 8
#define H 1024
#define W 1024
#define TILE 16
#define PP 10000
#define SIGMA_C 0.03f
#define MIN_D (-0.001f)
#define MAX_D 80.0f
#define PI_F 3.14159265358979323846f

typedef __attribute__((ext_vector_type(16))) _Float16 v16h;
typedef __attribute__((ext_vector_type(8)))  float    v8f;
typedef unsigned int uint4e __attribute__((ext_vector_type(4)));
typedef int          int4e  __attribute__((ext_vector_type(4)));
typedef int          int8e  __attribute__((ext_vector_type(8)));

#if defined(__HIP_DEVICE_COMPILE__) && __has_builtin(__builtin_amdgcn_tensor_load_to_lds)
#define USE_TDM 1
#else
#define USE_TDM 0
#endif

// ---------------- deterministic hash RNG ----------------
__device__ __forceinline__ unsigned int mix32(unsigned int x) {
    x ^= x >> 16; x *= 0x7feb352du;
    x ^= x >> 15; x *= 0x846ca68bu;
    x ^= x >> 16; return x;
}
__device__ __forceinline__ unsigned int rng3(unsigned int a, unsigned int b, unsigned int c) {
    return mix32(a * 0x9E3779B1u ^ mix32(b * 0x85EBCA77u ^ mix32(c + 0x165667B1u)));
}

// ---------------- K0: zero accumulators ----------------
__global__ void k_init(float* ws) {
    if (threadIdx.x < 32) ws[threadIdx.x] = 0.0f;   // edgemax[8], maxdiff[8], losssum[8], cnt[8]
}

// ---------------- K1: per-image edge max via TDM feed + WMMA Sobel ----------------
// One wave = one 16x16 output tile. Sobel is separable:
//   gx = Vsmooth(16x18 band) x Hdiff(18x16)   -> one v_wmma_f32_16x16x32_f16 (K padded 18->32)
//   gy = Vdiff  (16x18 band) x Hsmooth(18x16) -> one v_wmma_f32_16x16x32_f16
// Patch (18x18 f32) is DMA'd into LDS by the Tensor Data Mover when available.
__global__ __launch_bounds__(256) void k_edge_max(const float* __restrict__ images,
                                                  unsigned int* __restrict__ edgemax_bits) {
    __shared__ float patch[8][18 * 18];
    const int lane   = threadIdx.x & 31;
    const int waveIn = threadIdx.x >> 5;
    const int waveId = blockIdx.x * 8 + waveIn;
    const int tilesPerImg = (W / TILE) * (H / TILE);          // 4096
    const int img = waveId / tilesPerImg;
    const int t   = waveId % tilesPerImg;
    const int ty  = t / (W / TILE);
    const int tx  = t % (W / TILE);
    const int row0 = ty * TILE - 1;
    const int col0 = tx * TILE - 1;
    const float* base = images + (size_t)img * 3u * H * W;    // channel 0
    float* lp = patch[waveIn];

#if USE_TDM
    {
        // 2-D TDM descriptor: tensor = 1024x1024 f32 image plane, tile = 18x18 (clipped at the
        // low edge by shifting the LDS target; high-edge OOB reads return zeros per ISA, and the
        // unwritten pad row/col only feeds border outputs that are masked before the reduction).
        const int gr0 = max(row0, 0), gc0 = max(col0, 0);
        const int sh_r = gr0 - row0, sh_c = gc0 - col0;       // 0 or 1
        const int tile_h = 18 - sh_r;
        const int tile_w = 18 - sh_c;
        unsigned lds_addr = (unsigned)(uintptr_t)lp + (unsigned)((sh_r * 18 + sh_c) * 4);
        unsigned long long ga = (unsigned long long)(uintptr_t)(base + (size_t)gr0 * W + gc0);

        uint4e g0;
        g0[0] = 1u;                                           // count=1, user descriptor
        g0[1] = lds_addr;                                     // D#.lds_addr
        g0[2] = (unsigned)(ga & 0xFFFFFFFFu);                 // global_addr[31:0]
        g0[3] = (unsigned)((ga >> 32) & 0x01FFFFFFu) | 0x80000000u; // addr[56:32] | type=2

        int8e g1 = {};
        g1[0] = (int)(2u << 16);                              // data_size = 4 bytes
        g1[1] = (int)(((unsigned)W & 0xFFFFu) << 16);         // tensor_dim0[15:0]
        g1[2] = (int)((((unsigned)W >> 16) & 0xFFFFu) | (((unsigned)H & 0xFFFFu) << 16));
        g1[3] = (int)((((unsigned)H >> 16) & 0xFFFFu) | ((unsigned)tile_w << 16)); // tile_dim0
        g1[4] = (int)((unsigned)tile_h & 0xFFFFu);            // tile_dim1
        g1[5] = (int)W;                                       // tensor_dim0_stride[31:0]
        int4e g2 = {}, g3 = {};
#if __clang_major__ >= 23
        int8e g4 = {};
        __builtin_amdgcn_tensor_load_to_lds(g0, g1, g2, g3, g4, 0);
#else
        __builtin_amdgcn_tensor_load_to_lds(g0, g1, g2, g3, 0);
#endif
#if __has_builtin(__builtin_amdgcn_s_wait_tensorcnt)
        __builtin_amdgcn_s_wait_tensorcnt(0);
#else
        asm volatile("s_wait_tensorcnt 0x0" ::: "memory");
#endif
    }
#else
    for (int q = lane; q < 18 * 18; q += 32) {
        int r = q / 18, c = q - r * 18;
        int gr = min(max(row0 + r, 0), H - 1);
        int gc = min(max(col0 + c, 0), W - 1);
        lp[q] = base[gr * W + gc];
    }
#endif
    __syncthreads();

    // Constant banded A operands (depend only on lane).
    // A 16x32, contiguous k order used here (same as B): lane L -> M = L&15; half e -> K = ((L>=16)?16:0)+e
    const int m     = lane & 15;
    const int half8 = (lane >> 4) << 3;
    v16h a_vs{}, a_vd{};
#pragma unroll
    for (int e = 0; e < 16; ++e) {
        int k = (half8 << 1) | e;
        int d = k - m;
        float vs = (d == 0) ? 1.f : (d == 1) ? 2.f : (d == 2) ? 1.f : 0.f;
        float vd = (d == 0) ? 1.f : (d == 2) ? -1.f : 0.f;
        a_vs[e] = (_Float16)vs;
        a_vd[e] = (_Float16)vd;
    }

    // B operands from LDS. B 32x16, contiguous k order used here (same as A): lane L -> N = L&15; half e -> K = ((L>=16)?16:0)+e
    const int n     = lane & 15;
    const int kbase = (lane >> 4) << 4;
    v16h b_hd{}, b_hs{};
#pragma unroll
    for (int e = 0; e < 16; ++e) {
        int r = kbase + e;                                    // patch row (K)
        float hd = 0.f, hs = 0.f;
        if (r < 18) {
            float p0 = lp[r * 18 + n];
            float p1 = lp[r * 18 + n + 1];
            float p2 = lp[r * 18 + n + 2];
            hd = p2 - p0;
            hs = p0 + 2.f * p1 + p2;
        }
        b_hd[e] = (_Float16)hd;
        b_hs[e] = (_Float16)hs;
    }

    v8f zero{};
    v8f gx = __builtin_amdgcn_wmma_f32_16x16x32_f16(false, a_vs, false, b_hd, (short)0, zero, false, false);
    v8f gy = __builtin_amdgcn_wmma_f32_16x16x32_f16(false, a_vd, false, b_hs, (short)0, zero, false, false);

    // max(sqrt(e_i)) == sqrt(max(e_i)): reduce squared magnitudes branch-free, one sqrt per wave.
    float mx2 = 0.f;
#pragma unroll
    for (int v = 0; v < 8; ++v) {                             // D layout: M = v + 8*(lane>=16), N = lane&15
        int M = v + ((lane >> 4) << 3);
        int Y = ty * TILE + M;
        int X = tx * TILE + n;
        bool interior = (Y >= 1) & (Y <= H - 2) & (X >= 1) & (X <= W - 2);
        float e2 = gx[v] * gx[v] + gy[v] * gy[v];
        mx2 = fmaxf(mx2, interior ? e2 : 0.f);
    }
#pragma unroll
    for (int off = 16; off >= 1; off >>= 1) mx2 = fmaxf(mx2, __shfl_xor(mx2, off, 32));
    if (lane == 0) atomicMax(&edgemax_bits[img], __float_as_uint(sqrtf(mx2)));
}

// ---------------- Sobel at a single point (recompute; no stored edge/theta maps) ----------------
__device__ __forceinline__ void sobel_at(const float* __restrict__ base, int y, int x,
                                         float& gx, float& gy) {
    float p00 = base[(y - 1) * W + x - 1], p01 = base[(y - 1) * W + x], p02 = base[(y - 1) * W + x + 1];
    float p10 = base[y * W + x - 1],                                    p12 = base[y * W + x + 1];
    float p20 = base[(y + 1) * W + x - 1], p21 = base[(y + 1) * W + x], p22 = base[(y + 1) * W + x + 1];
    gx = (p02 - p00) + 2.f * (p12 - p10) + (p22 - p20);
    gy = (p00 + 2.f * p01 + p02) - (p20 + 2.f * p21 + p22);
}

// ---------------- K2: rejection-sample edge points, build pair data, maxdiff ----------------
__global__ void k_sample(const float* __restrict__ inputs, const float* __restrict__ targets,
                         const float* __restrict__ images, const float* __restrict__ depth_gt,
                         const unsigned int* __restrict__ edgemax_bits,
                         unsigned int* __restrict__ maxdiff_bits,
                         float* __restrict__ pair_dI, float* __restrict__ pair_tA,
                         float* __restrict__ pair_tB, float* __restrict__ pair_vm) {
    int s   = blockIdx.x * blockDim.x + threadIdx.x;
    int img = blockIdx.y;
    if (s >= PP) return;

    const size_t off1 = (size_t)img * H * W;
    const float* imbase = images + (size_t)img * 3u * H * W;  // channel 0
    const float  emax   = __uint_as_float(edgemax_bits[img]);
    const float  ethr2  = 0.01f * emax * emax;                // (0.1*max)^2, edge>=0 -> compare squares

    // uniform-from-masked-set == categorical with uniform p over the mask (reference semantics)
    int sy = 1, sx = 1;
    float gx = 0.f, gy = 0.f;
    for (int t = 0; t < 16; ++t) {
        unsigned int h = rng3((unsigned)img, (unsigned)s, (unsigned)t);
        unsigned int idx = h % (1022u * 1022u);
        int y = 1 + (int)(idx / 1022u);
        int x = 1 + (int)(idx % 1022u);
        float gxx, gyy;
        sobel_at(imbase, y, x, gxx, gyy);
        float ev2 = gxx * gxx + gyy * gyy;
        float d  = depth_gt[off1 + (size_t)y * W + x];
        float tv = targets[off1 + (size_t)y * W + x];
        sy = y; sx = x; gx = gxx; gy = gyy;
        bool ok = (ev2 >= ethr2) & (d > MIN_D) & (d < MAX_D) & (tv != 80.0f);
        if (ok) break;
    }

    float th  = atan2f(gy, gx);
    float a   = th + 0.5f * PI_F + PI_F;
    a = fmodf(a, 2.f * PI_F); if (a < 0.f) a += 2.f * PI_F;
    float th2 = a - PI_F;
    bool pdir = (mix32(0x5EED1234u ^ (unsigned)img) & 1u) == 0u;

    float cth, sth; __sincosf(th,  &sth, &cth);
    float cth2, sth2; __sincosf(th2, &sth2, &cth2);

    int row[4], col[4];
    bool valid = true;
#pragma unroll
    for (int j = 0; j < 4; ++j) {
        unsigned int h = rng3((unsigned)img, (unsigned)s, 16u + (unsigned)j);
        float dist = (float)(2 + (int)(h % 29u));             // randint(2,31) -> [2,30]
        dist *= (j < 2) ? -1.f : 1.f;
        int c1 = sx + (int)rintf(dist * cth);
        int r1 = sy + (int)rintf(dist * sth);
        int c2 = sx + (int)rintf(dist * sth2);
        int r2 = sy + (int)rintf(dist * cth2);
        int c = pdir ? c1 : c2;
        int r = pdir ? r1 : r2;
        valid = valid & (c >= 0) & (c <= W - 1) & (r >= 0) & (r <= H - 1);
        col[j] = min(max(c, 0), W - 1);
        row[j] = min(max(r, 0), H - 1);
    }
    float vm = valid ? 1.f : 0.f;

    float tv[4], iv[4];
#pragma unroll
    for (int j = 0; j < 4; ++j) {
        size_t p = off1 + (size_t)row[j] * W + col[j];
        tv[j] = targets[p];
        iv[j] = inputs[p];
    }

    float mdl = 0.f;
#pragma unroll
    for (int p = 0; p < 3; ++p) {                             // pairs (0,1),(1,2),(2,3)
        float tA = tv[p], tB = tv[p + 1];
        size_t j = ((size_t)img * 3 + p) * PP + s;
        pair_tA[j] = tA;
        pair_tB[j] = tB;
        pair_dI[j] = iv[p] - iv[p + 1];
        pair_vm[j] = vm;
        if (valid) mdl = fmaxf(mdl, fabsf(tA - tB));
    }
    if (mdl > 0.f) atomicMax(&maxdiff_bits[img], __float_as_uint(mdl));
}

// ---------------- K3: per-pair ranking loss + reduction ----------------
__global__ void k_loss(const float* __restrict__ pair_dI, const float* __restrict__ pair_tA,
                       const float* __restrict__ pair_tB, const float* __restrict__ pair_vm,
                       const unsigned int* __restrict__ maxdiff_bits,
                       float* __restrict__ losssum, float* __restrict__ cntsum) {
    int j   = blockIdx.x * blockDim.x + threadIdx.x;
    int img = blockIdx.y;
    float l = 0.f, c = 0.f;
    if (j < 3 * PP) {
        size_t o = (size_t)img * 3 * PP + j;
        float tA = pair_tA[o], tB = pair_tB[o], dI = pair_dI[o], vm = pair_vm[o];
        float maxd  = __uint_as_float(maxdiff_bits[img]);
        float ratio = (tA + 1e-6f) / (tB + 1e-6f);
        float absd  = fabsf(tA - tB);
        float tw    = __expf(absd / (maxd + 1e-6f));
        float hi = 1.f + SIGMA_C, lo = 1.f / (1.f + SIGMA_C);
        bool  eq    = (ratio < hi) & (ratio > lo);
        float label = (ratio >= hi) ? 1.f : ((ratio <= lo) ? -1.f : 0.f);
        float eqf   = eq ? 1.f : 0.f;
        float el    = dI * dI / tw * eqf;                     // (iA-iB)^2 / tw * eq
        float ul    = log1pf(__expf(-dI * label)) * (1.f - eqf); // (iB-iA)*labels
        l = (el + ul) * vm;                                   // ALPHA == 1
        c = vm;
    }
#pragma unroll
    for (int off = 16; off >= 1; off >>= 1) {
        l += __shfl_xor(l, off, 32);
        c += __shfl_xor(c, off, 32);
    }
    if ((threadIdx.x & 31) == 0) {
        atomicAdd(&losssum[img], l);
        atomicAdd(&cntsum[img], c);
    }
}

// ---------------- K4: finalize the two scalars ----------------
__global__ void k_final(const float* __restrict__ losssum, const float* __restrict__ cntsum,
                        float* __restrict__ out) {
    if (threadIdx.x == 0) {
        float L = 0.f, C = 0.f;
        for (int i = 0; i < NIMG; ++i) {
            float c = cntsum[i];
            L += losssum[i] / fmaxf(c, 1.f);
            C += c;
        }
        out[0] = L / (float)NIMG;
        out[1] = C / (float)NIMG;
    }
}

extern "C" void kernel_launch(void* const* d_in, const int* in_sizes, int n_in,
                              void* d_out, int out_size, void* d_ws, size_t ws_size,
                              hipStream_t stream) {
    const float* inputs   = (const float*)d_in[0];
    const float* targets  = (const float*)d_in[1];
    const float* images   = (const float*)d_in[2];
    const float* depth_gt = (const float*)d_in[3];
    float* out = (float*)d_out;

    float*        ws       = (float*)d_ws;
    unsigned int* edgemaxB = (unsigned int*)ws;               // [0..7]
    unsigned int* maxdiffB = (unsigned int*)(ws + 8);         // [8..15]
    float*        losssum  = ws + 16;                         // [16..23]
    float*        cntsum   = ws + 24;                         // [24..31]
    const size_t  PAIRS = (size_t)NIMG * 3 * PP;              // 240000
    float* pair_dI = ws + 32;
    float* pair_tA = pair_dI + PAIRS;
    float* pair_tB = pair_tA + PAIRS;
    float* pair_vm = pair_tB + PAIRS;

    k_init<<<1, 32, 0, stream>>>(ws);

    const int tiles = NIMG * (W / TILE) * (H / TILE);         // 32768 waves
    k_edge_max<<<tiles / 8, 256, 0, stream>>>(images, edgemaxB);

    dim3 g2((PP + 255) / 256, NIMG);
    k_sample<<<g2, 256, 0, stream>>>(inputs, targets, images, depth_gt,
                                     edgemaxB, maxdiffB, pair_dI, pair_tA, pair_tB, pair_vm);

    dim3 g3((3 * PP + 255) / 256, NIMG);
    k_loss<<<g3, 256, 0, stream>>>(pair_dI, pair_tA, pair_tB, pair_vm,
                                   maxdiffB, losssum, cntsum);

    k_final<<<1, 32, 0, stream>>>(losssum, cntsum, out);
}
